// SparseConv2D_40200893890938
// MI455X (gfx1250) — matrix operand order repaired and checked by reading, not compiled, not run on hardware
//
#include <hip/hip_runtime.h>

typedef __attribute__((ext_vector_type(16))) __bf16 v16bf;
typedef __attribute__((ext_vector_type(8)))  __bf16 v8bf;
typedef __attribute__((ext_vector_type(8)))  float  v8f;

#define KOUT 126
#define CIN 128
#define HW 112
#define PIX_PER_IMG (HW * HW)          /* 12544 */
#define NIMG 32
#define TOTAL_PIX (NIMG * PIX_PER_IMG) /* 401408 */
#define KTOT (9 * CIN)                 /* 1152 */
#define NKC (KTOT / 32)                /* 36 K-chunks of (tap, 32 channels) */
#define NTILES 8                       /* 8 x 16 = 128 output channels (126 used) */
#define MT 128                         /* pixels per block */
#define SUBT 8                         /* M-subtiles of 16 per wave */
#define ABUF_ELEMS (SUBT * 32 * 16)    /* one A slab: 128px x 32ch bf16 = 8 KB */

// fp32 -> bf16 RNE for the (tiny) weight-pack kernel
__device__ __forceinline__ __bf16 f2bf(float f) {
    union { float f; unsigned u; } uf; uf.f = f;
    unsigned u = uf.u;
    unsigned r = (u + 0x7FFFu + ((u >> 16) & 1u)) >> 16;
    unsigned short h = (unsigned short)r;
    union { unsigned short s; __bf16 b; } ub; ub.s = h;
    return ub.b;
}

// Scatter sparse weights into dense bf16 B, pre-packed in WMMA-B per-lane layout:
// element t = ((kc*8 + ntile)*32 + lane)*16 + j holds
//   W[K = kc*32 + (lane<16 ? j : 16+j)][N = ntile*16 + lane%16],  K = tap*128 + cin.
__global__ void __launch_bounds__(256)
pack_weights_kernel(const float* __restrict__ sk, const int* __restrict__ pat,
                    __bf16* __restrict__ Bp) {
    int t = blockIdx.x * 256 + threadIdx.x;
    if (t >= NKC * NTILES * 32 * 16) return;
    int j  = t & 15;
    int l  = (t >> 4) & 31;
    int nt = (t >> 9) & 7;
    int kc = t >> 12;
    int n  = nt * 16 + (l & 15);
    int klocal = (l < 16) ? j : (16 + j);
    int k = kc * 32 + klocal;
    int p = k >> 7;    // tap id 0..8
    int c = k & 127;   // input channel
    float w = 0.0f;
    if (n < KOUT) {
#pragma unroll
        for (int i = 0; i < 5; ++i) {
            if (pat[n * 5 + i] == p) { w = sk[(i * CIN + c) * KOUT + n]; }
        }
    }
    Bp[t] = f2bf(w);
}

// Implicit-GEMM 3x3 conv. Block = 128 pixels x 128 channels, 8 waves.
// Wave wv owns N-tile wv (16 ch) and 8 M-subtiles (64 f32 accum VGPRs).
// Double-buffered LDS A slab -> one barrier per K-chunk.
__global__ void __launch_bounds__(256)
sparse_conv_wmma_kernel(const float* __restrict__ in, const __bf16* __restrict__ Bp,
                        const float* __restrict__ bias, float* __restrict__ out) {
    __shared__ __align__(32) __bf16 Abuf[2 * ABUF_ELEMS]; // 16 KB double buffer

    const int t    = threadIdx.x;
    const int lane = t & 31;
    const int wv   = t >> 5;

    const int P   = blockIdx.x * MT;
    const int img = P / PIX_PER_IMG;
    const int rem = P - img * PIX_PER_IMG;   // P multiple of 128; 12544/128=98 -> tile within one image

    // Producer: each thread stages two 8-channel slots per chunk (slot t and t+256).
    const int g  = t & 3;                    // 8-ch group within the 32-ch chunk
    const int m0 = t >> 2;                   // pixel 0..63
    const int m1 = m0 + 64;                  // pixel 64..127
    const int h0 = (rem + m0) / HW, w0 = (rem + m0) - h0 * HW;
    const int h1 = (rem + m1) / HW, w1 = (rem + m1) - h1 * HW;
    // A lane layout: lanes 0-15 hold K {0-15}; lanes 16-31 hold K {16-31}
    const int lds0 = (m0 >> 4) * 512 + (((m0 & 15) + ((g >> 1) << 4)) << 4) + ((g & 1) << 3);
    const int lds1 = (m1 >> 4) * 512 + (((m1 & 15) + ((g >> 1) << 4)) << 4) + ((g & 1) << 3);
    const long gb0 = ((long)(img * PIX_PER_IMG + h0 * HW + w0)) * CIN + g * 8;
    const long gb1 = ((long)(img * PIX_PER_IMG + h1 * HW + w1)) * CIN + g * 8;

    v8f acc[SUBT] = {};
    v8f r0, r1;

    auto load_chunk = [&](int kc) {
        const int p  = kc >> 2;
        const int cb = kc & 3;
        const int dy = p / 3 - 1;
        const int dx = p - (p / 3) * 3 - 1;
        const long off = (long)(dy * HW + dx) * CIN + cb * 32;
        const v8f z = {};
        bool va = ((unsigned)(h0 + dy) < HW) && ((unsigned)(w0 + dx) < HW);
        bool vb = ((unsigned)(h1 + dy) < HW) && ((unsigned)(w1 + dx) < HW);
        r0 = va ? *(const v8f*)(in + gb0 + off) : z;   // 2x global_load_b128
        r1 = vb ? *(const v8f*)(in + gb1 + off) : z;
    };
    auto store_chunk = [&](int boff) {
        *(v8bf*)&Abuf[boff + lds0] = __builtin_convertvector(r0, v8bf); // v_cvt_pk_bf16_f32 + ds_store_b128
        *(v8bf*)&Abuf[boff + lds1] = __builtin_convertvector(r1, v8bf);
    };

    load_chunk(0);
    store_chunk(0);

    for (int kc = 0; kc < NKC; ++kc) {
        __syncthreads();                       // chunk kc visible; prior reads of other buffer done
        if (kc + 1 < NKC) load_chunk(kc + 1);  // overlap global latency with WMMAs

        const int boff = (kc & 1) * ABUF_ELEMS;
        // B operand: 32B/lane from pre-packed weights (288 KB total, L2/WGP$ resident)
        v16bf b = *(const v16bf*)(Bp + (((kc * NTILES + wv) * 32 + lane) << 4));

#pragma unroll
        for (int s = 0; s < SUBT; ++s) {
            v16bf a = *(const v16bf*)&Abuf[boff + s * 512 + (lane << 4)];
            acc[s] = __builtin_amdgcn_wmma_f32_16x16x32_bf16(
                false, a, false, b, (short)0, acc[s], false, false);
        }

        if (kc + 1 < NKC) store_chunk(boff ^ ABUF_ELEMS); // fill the other buffer
    }

    // Epilogue: C/D layout VGPR i -> M = i + (lane<16 ? 0 : 8), N = lane%16
    const int nloc = lane & 15;
    const int ch   = wv * 16 + nloc;
    const int rb   = (lane < 16) ? 0 : 8;
    if (ch < KOUT) {
        const float bz = bias[ch];
#pragma unroll
        for (int s = 0; s < SUBT; ++s) {
#pragma unroll
            for (int i = 0; i < 8; ++i) {
                int pix = P + s * 16 + rb + i;
                float v = acc[s][i] + bz;
                v = v > 0.0f ? v : 0.0f;
                // streaming output: non-temporal so it doesn't evict the activation reuse set
                __builtin_nontemporal_store(v, &out[(long)pix * KOUT + ch]);
            }
        }
    }
}

extern "C" void kernel_launch(void* const* d_in, const int* in_sizes, int n_in,
                              void* d_out, int out_size, void* d_ws, size_t ws_size,
                              hipStream_t stream) {
    const float* inp  = (const float*)d_in[0];   // (32,112,112,128) f32
    const float* sk   = (const float*)d_in[1];   // (5,128,126) f32
    const float* bias = (const float*)d_in[2];   // (126,) f32
    const int*   pat  = (const int*)d_in[3];     // (126,5) i32
    float*       outp = (float*)d_out;           // (32,112,112,126) f32
    __bf16*      Bp   = (__bf16*)d_ws;           // 294912 B packed weights

    const int packElems = NKC * NTILES * 32 * 16;            // 147456
    pack_weights_kernel<<<dim3((packElems + 255) / 256), dim3(256), 0, stream>>>(sk, pat, Bp);
    sparse_conv_wmma_kernel<<<dim3(TOTAL_PIX / MT), dim3(256), 0, stream>>>(inp, Bp, bias, outp);
}
